// PoemGenerator_26946624815198
// MI455X (gfx1250) — matrix operand order repaired and checked by reading, not compiled, not run on hardware
//
#include <hip/hip_runtime.h>

// ---------------------------------------------------------------------------
// Problem constants (from reference): B=16, L=128, V=32000, E=512, H=1024, D=512
// ---------------------------------------------------------------------------
constexpr int Bc   = 16;
constexpr int Lc   = 128;
constexpr int Vc   = 32000;
constexpr int Ec   = 512;
constexpr int Hc   = 1024;
constexpr int Dc   = 512;
constexpr int EPD  = Ec + Dc;    // 1024 (x row length)
constexpr int G4H  = 4 * Hc;     // 4096
constexpr int NSTEP = Lc - 1;    // 127 recurrent steps
constexpr int TM   = 4;          // timestep (M) blocking in logits GEMM

// ---------------------------------------------------------------------------
// Vector types for WMMA (wave32, gfx1250)
// ---------------------------------------------------------------------------
typedef __attribute__((ext_vector_type(16))) __bf16         bf16x16;
typedef __attribute__((ext_vector_type(8)))  float          f32x8;
typedef __attribute__((ext_vector_type(8)))  unsigned short u16x8;

union Bf16Frag { u16x8 u[2]; bf16x16 b; };

// f32 -> bf16 round-to-nearest-even (manual, stored as u16)
__device__ __forceinline__ unsigned short f2bf(float f) {
    unsigned int u = __float_as_uint(f);
    u = (u + 0x7FFFu + ((u >> 16) & 1u)) >> 16;
    return (unsigned short)u;
}

__device__ __forceinline__ float sigmoidf_(float x) {
    return 1.0f / (1.0f + __expf(-x));
}

// A fragment: 16x32 bf16 tile from row-major A (lda elems), tile base already
// offset to current k. ISA layout: lanes 0-15 hold K[0..8)&[16..24) of row m,
// lanes 16-31 hold K[8..16)&[24..32).
__device__ __forceinline__ bf16x16 frag_a(const unsigned short* tile, int lda) {
    int lane = threadIdx.x & 31;
    int m  = lane & 15;
    int kh = lane >> 4;             // 0 or 1
    const unsigned short* p = tile + m * lda + kh * 8;
    Bf16Frag f;
    f.u[0] = *(const u16x8*)(p);
    f.u[1] = *(const u16x8*)(p + 16);
    return f.b;
}

// B fragment: 32x16 bf16 tile where column n of B is row n of a row-major
// (N,K) weight matrix. Lane holds K values of weight row n:
// lanes 0-15 -> K[0..8)&[16..24), lanes 16-31 -> K[8..16)&[24..32).
__device__ __forceinline__ bf16x16 frag_b(const unsigned short* wrow, int ldb) {
    int lane = threadIdx.x & 31;
    int n  = lane & 15;
    int kh = lane >> 4;
    const unsigned short* p = wrow + n * ldb + kh * 8;
    Bf16Frag f;
    f.u[0] = *(const u16x8*)(p);
    f.u[1] = *(const u16x8*)(p + 16);
    return f.b;
}

// ---------------------------------------------------------------------------
// Prep kernels
// ---------------------------------------------------------------------------
__global__ void k_cvt_bf16(const float* __restrict__ src,
                           unsigned short* __restrict__ dst, int n) {
    for (int i = blockIdx.x * blockDim.x + threadIdx.x; i < n;
         i += gridDim.x * blockDim.x)
        dst[i] = f2bf(src[i]);
}

__global__ void k_bias(const float* __restrict__ bi, const float* __restrict__ bh,
                       float* __restrict__ bias) {
    int i = blockIdx.x * blockDim.x + threadIdx.x;
    if (i < G4H) bias[i] = bi[i] + bh[i];
}

// x[s][b][k] = k<E ? emb[target[b,s]][k] : latent[b][k-E]   (bf16)
__global__ void k_build_x(const float* __restrict__ emb,
                          const float* __restrict__ latent,
                          const int* __restrict__ target,
                          unsigned short* __restrict__ x) {
    int total = NSTEP * Bc * EPD;
    for (int idx = blockIdx.x * blockDim.x + threadIdx.x; idx < total;
         idx += gridDim.x * blockDim.x) {
        int k  = idx & (EPD - 1);
        int rb = idx >> 10;         // EPD == 1024
        int b  = rb & 15;
        int s  = rb >> 4;
        float v;
        if (k < Ec) v = emb[(long)target[b * Lc + s] * Ec + k];
        else        v = latent[b * Dc + (k - Ec)];
        x[idx] = f2bf(v);
    }
}

// zero out[:,0,:], h_seq slot 0 + padded tail slots, c state
// (re-done every launch: no cross-call state)
__global__ void k_init_zero(float* __restrict__ out,
                            unsigned short* __restrict__ hseq,
                            float* __restrict__ c0) {
    int idx = blockIdx.x * blockDim.x + threadIdx.x;
    int tot = Bc * Vc;
    if (idx < tot) {
        int b = idx / Vc, v = idx - b * Vc;
        out[(long)b * Lc * Vc + v] = 0.0f;
    }
    if (idx < Bc * Hc) { hseq[idx] = 0; c0[idx] = 0.0f; }
    // zero TM padded slots past t=Lc-1 so the M-blocked logits GEMM can load
    // them unguarded
    if (idx < TM * Bc * Hc) hseq[(size_t)Lc * Bc * Hc + idx] = 0;
}

// ---------------------------------------------------------------------------
// GEMM 1: Gx[s] = x[s] @ W_ih^T + bias     (M=16 per step, N=4096, K=1024)
// grid (64, 127), block 128 (4 waves, one 16x16 N-tile each)
// ---------------------------------------------------------------------------
__global__ __launch_bounds__(128)
void k_gx_gemm(const unsigned short* __restrict__ x,
               const unsigned short* __restrict__ wih,
               const float* __restrict__ bias,
               float* __restrict__ gx) {
    int wave  = threadIdx.x >> 5;
    int lane  = threadIdx.x & 31;
    int ntile = blockIdx.x * 4 + wave;          // 0..255
    int s     = blockIdx.y;                     // 0..126
    const unsigned short* A  = x   + s * Bc * EPD;
    const unsigned short* Bw = wih + (long)ntile * 16 * EPD;
    f32x8 acc = {};
#pragma unroll 4
    for (int kk = 0; kk < EPD; kk += 32) {
        bf16x16 a = frag_a(A + kk, EPD);
        bf16x16 b = frag_b(Bw + kk, EPD);
        acc = __builtin_amdgcn_wmma_f32_16x16x32_bf16(false, a, false, b,
                                                      (short)0, acc, false, false);
    }
    int n = lane & 15, mb = (lane >> 4) * 8;
    float bv = bias[ntile * 16 + n];
    float* O = gx + (long)s * Bc * G4H + ntile * 16;
#pragma unroll
    for (int i = 0; i < 8; ++i) O[(mb + i) * G4H + n] = acc[i] + bv;
}

// ---------------------------------------------------------------------------
// Recurrent step: gates = Gx[s] + h_prev @ W_hh^T; LSTM elementwise.
// grid 64 (hidden tiles of 16), block 128: wave g computes gate g's tile.
// ---------------------------------------------------------------------------
__global__ __launch_bounds__(128)
void k_lstm_step(const unsigned short* __restrict__ hprev,
                 const unsigned short* __restrict__ whh,
                 const float* __restrict__ gx_s,
                 float* __restrict__ c,
                 unsigned short* __restrict__ hnext) {
    __shared__ float gsm[4][16][16];
    int wave = threadIdx.x >> 5;
    int lane = threadIdx.x & 31;
    int hx   = blockIdx.x;                     // hidden tile 0..63
    int j0   = wave * Hc + hx * 16;            // gate-row base in [0,4096)
    const unsigned short* Bw = whh + (long)j0 * Hc;
    f32x8 acc = {};
#pragma unroll 4
    for (int kk = 0; kk < Hc; kk += 32) {
        bf16x16 a = frag_a(hprev + kk, Hc);
        bf16x16 b = frag_b(Bw + kk, Hc);
        acc = __builtin_amdgcn_wmma_f32_16x16x32_bf16(false, a, false, b,
                                                      (short)0, acc, false, false);
    }
    int n = lane & 15, mb = (lane >> 4) * 8;
#pragma unroll
    for (int i = 0; i < 8; ++i)
        gsm[wave][mb + i][n] = acc[i] + gx_s[(mb + i) * G4H + j0 + n];
    __syncthreads();
    for (int e = threadIdx.x; e < 256; e += 128) {
        int m = e >> 4, n2 = e & 15;
        float ig = sigmoidf_(gsm[0][m][n2]);
        float fg = sigmoidf_(gsm[1][m][n2]);
        float gg = tanhf   (gsm[2][m][n2]);
        float og = sigmoidf_(gsm[3][m][n2]);
        int col  = hx * 16 + n2;
        float cn = fg * c[m * Hc + col] + ig * gg;
        c[m * Hc + col] = cn;
        hnext[m * Hc + col] = f2bf(og * tanhf(cn));
    }
}

// ---------------------------------------------------------------------------
// GEMM 2 (dominant): logits[t] = h[t] @ fc_w^T + fc_b for t=1..127.
// M-blocked over TM=4 timesteps per wave: each fc_w (B) fragment is loaded
// once and reused for 4 WMMAs, cutting fc_w L2 traffic 4x.
// grid (250, ceil(127/TM)=32), block 256 (8 waves, one 16-wide N-tile each).
// ---------------------------------------------------------------------------
__global__ __launch_bounds__(256)
void k_logits(const unsigned short* __restrict__ hseq,
              const unsigned short* __restrict__ fcw,
              const float* __restrict__ fcb,
              float* __restrict__ out) {
    int wave  = threadIdx.x >> 5;
    int lane  = threadIdx.x & 31;
    int ntile = blockIdx.x * 8 + wave;          // 0..1999
    int t0    = 1 + blockIdx.y * TM;            // 1,5,...,125
    const unsigned short* Bw = fcw  + (long)ntile * 16 * Hc;
    const unsigned short* A0 = hseq + (long)t0 * Bc * Hc;

    f32x8 acc[TM];
#pragma unroll
    for (int m = 0; m < TM; ++m) acc[m] = (f32x8){};

#pragma unroll 2
    for (int kk = 0; kk < Hc; kk += 32) {
        __builtin_prefetch((const void*)(Bw + (lane & 15) * Hc + kk + 128), 0, 0);
        bf16x16 b = frag_b(Bw + kk, Hc);
#pragma unroll
        for (int m = 0; m < TM; ++m) {
            bf16x16 a = frag_a(A0 + (long)m * Bc * Hc + kk, Hc);
            acc[m] = __builtin_amdgcn_wmma_f32_16x16x32_bf16(
                false, a, false, b, (short)0, acc[m], false, false);
        }
    }

    int n = lane & 15, mb = (lane >> 4) * 8;
    int col = ntile * 16 + n;
    float bv = fcb[col];
#pragma unroll
    for (int m = 0; m < TM; ++m) {
        int t = t0 + m;
        if (t < Lc) {
#pragma unroll
            for (int i = 0; i < 8; ++i) {
                int bidx = mb + i;
                out[(long)bidx * Lc * Vc + (long)t * Vc + col] = acc[m][i] + bv;
            }
        }
    }
}

// ---------------------------------------------------------------------------
// Host-side launch (graph-capturable: only kernel launches on `stream`)
// ---------------------------------------------------------------------------
extern "C" void kernel_launch(void* const* d_in, const int* in_sizes, int n_in,
                              void* d_out, int out_size, void* d_ws, size_t ws_size,
                              hipStream_t stream) {
    const float* latent = (const float*)d_in[0];
    const float* emb    = (const float*)d_in[1];
    const float* W_ih   = (const float*)d_in[2];
    const float* W_hh   = (const float*)d_in[3];
    const float* b_ih   = (const float*)d_in[4];
    const float* b_hh   = (const float*)d_in[5];
    const float* fc_w   = (const float*)d_in[6];
    const float* fc_b   = (const float*)d_in[7];
    const int*   target = (const int*)d_in[8];
    float* out = (float*)d_out;

    // workspace bump allocator (256B aligned)
    char* p = (char*)d_ws;
    auto alloc = [&](size_t bytes) -> char* {
        char* r = p;
        p += (bytes + 255) & ~(size_t)255;
        return r;
    };
    unsigned short* wih_bf = (unsigned short*)alloc((size_t)G4H * EPD * 2);
    unsigned short* whh_bf = (unsigned short*)alloc((size_t)G4H * Hc  * 2);
    unsigned short* fcw_bf = (unsigned short*)alloc((size_t)Vc  * Hc  * 2);
    unsigned short* x_bf   = (unsigned short*)alloc((size_t)NSTEP * Bc * EPD * 2);
    unsigned short* h_seq  = (unsigned short*)alloc((size_t)(Lc + TM) * Bc * Hc * 2);
    float*          gx     = (float*)alloc((size_t)NSTEP * Bc * G4H * 4);
    float*          cbuf   = (float*)alloc((size_t)Bc * Hc * 4);
    float*          bias   = (float*)alloc((size_t)G4H * 4);

    // 1. weight conversion to bf16 (once per launch)
    k_cvt_bf16<<<1024, 256, 0, stream>>>(W_ih, wih_bf, G4H * EPD);
    k_cvt_bf16<<<1024, 256, 0, stream>>>(W_hh, whh_bf, G4H * Hc);
    k_cvt_bf16<<<4096, 256, 0, stream>>>(fc_w, fcw_bf, Vc * Hc);
    k_bias<<<(G4H + 255) / 256, 256, 0, stream>>>(b_ih, b_hh, bias);
    // 2. embedding gather + latent concat -> x_bf (teacher forcing: tokens known)
    k_build_x<<<2048, 256, 0, stream>>>(emb, latent, target, x_bf);
    // 3. zero out[:,0,:], h0 + padded h slots, c0
    k_init_zero<<<(Bc * Vc + 255) / 256, 256, 0, stream>>>(out, h_seq, cbuf);
    // 4. batched input-projection GEMM for all 127 steps
    k_gx_gemm<<<dim3(64, NSTEP), 128, 0, stream>>>(x_bf, wih_bf, bias, gx);
    // 5. sequential LSTM recurrence (only the h@W_hh^T part is serial)
    for (int s = 0; s < NSTEP; ++s) {
        k_lstm_step<<<64, 128, 0, stream>>>(
            h_seq + (size_t)s * Bc * Hc, whh_bf,
            gx + (size_t)s * Bc * G4H, cbuf,
            h_seq + (size_t)(s + 1) * Bc * Hc);
    }
    // 6. one big logits GEMM over all saved hidden states (dominant FLOPs)
    k_logits<<<dim3(250, (NSTEP + TM - 1) / TM), 256, 0, stream>>>(
        h_seq, fcw_bf, fc_b, out);
}
